// Decoder_8461085573653
// MI455X (gfx1250) — matrix operand order repaired and checked by reading, not compiled, not run on hardware
//
#include <hip/hip_runtime.h>
#include <stdint.h>

#define E_DIM 256
#define H_DIM 512
#define S_DIM 128
#define B_DIM 512
#define T_STEPS 100

typedef _Float16 h16v  __attribute__((ext_vector_type(16)));
typedef float    f32x8 __attribute__((ext_vector_type(8)));
typedef unsigned int u32x4 __attribute__((ext_vector_type(4)));
typedef unsigned int v4u __attribute__((ext_vector_type(4)));
typedef int v8i_t __attribute__((ext_vector_type(8)));
typedef int v4i_t __attribute__((ext_vector_type(4)));

__device__ __forceinline__ f32x8 wmma_f16(h16v a, h16v b, f32x8 c) {
  return __builtin_amdgcn_wmma_f32_16x16x32_f16(false, a, false, b, (short)0, c, false, false);
}

__device__ __forceinline__ float sigmoid_f(float x) { return 1.0f / (1.0f + __expf(-x)); }

// ---------------- Tensor Data Mover: 2D f16 tile -> LDS ----------------------
// D# per CDNA5 ISA ch.8: group0 = {count/flags, lds_addr, global_addr, type=2},
// group1 = {data_size=2B, tensor_dim0/1, tile_dim0/1, dim0 stride}. Groups 2/3
// and the trailing group are zeros (tile_dim2..4 = 0 -> unused).
// Toolchain builtin is the 6-arg form: (v4u, v8i, v4i, v4i, v8i, i32 cpol).
__device__ __forceinline__ void tdm_load_tile_2d(void* lds_dst, const void* gsrc,
                                                 uint32_t w_elems, uint32_t h_rows,
                                                 uint32_t stride_elems) {
  uint32_t lds_off = (uint32_t)(uintptr_t)lds_dst;  // flat LDS aperture: addr[31:0] is LDS byte offset
  uint64_t ga = (uint64_t)(uintptr_t)gsrc;
  v4u g0;
  g0[0] = 1u;                                            // count=1, user descriptor
  g0[1] = lds_off;                                       // lds_addr  (bits 63:32)
  g0[2] = (uint32_t)ga;                                  // global_addr[31:0] (bits 95:64)
  g0[3] = (uint32_t)((ga >> 32) & 0x01FFFFFFu) | (2u << 30); // addr[56:32] + type=2 (bits 127:126)
  v8i_t g1;
  g1[0] = (int)(1u << 16);                               // workgroup_mask=0, data_size=1 (2 bytes)
  g1[1] = (int)((w_elems & 0xFFFFu) << 16);              // tensor_dim0[15:0]  (bits 63:48)
  g1[2] = (int)((h_rows & 0xFFFFu) << 16);               // tensor_dim1[15:0]  (bits 95:80)
  g1[3] = (int)((w_elems & 0xFFFFu) << 16);              // tile_dim0          (bits 127:112)
  g1[4] = (int)(h_rows & 0xFFFFu);                       // tile_dim1 (bits 143:128), tile_dim2=0
  g1[5] = (int)stride_elems;                             // tensor_dim0_stride[31:0] (bits 191:160)
  g1[6] = 0;                                             // stride hi, dim1_stride lo
  g1[7] = 0;
  v4i_t gz4 = {0, 0, 0, 0};
  v8i_t gz8 = {0, 0, 0, 0, 0, 0, 0, 0};
  __builtin_amdgcn_tensor_load_to_lds(g0, g1, gz4, gz4, gz8, 0);
}

// ---------------- WMMA fragment loaders (CDNA5 wave32, ISA 7.12.2) -----------
__device__ __forceinline__ h16v frag_a16(const _Float16* __restrict__ A, int lda, int k0, int lane) {
  int m = lane & 15, g = (lane >> 4) & 1;
  const _Float16* p = A + (size_t)m * lda + (k0 + g * 8);
  union { h16v v; u32x4 q[2]; } u;
  u.q[0] = *(const u32x4*)(p);
  u.q[1] = *(const u32x4*)(p + 16);
  return u.v;
}
__device__ __forceinline__ h16v frag_b16(const _Float16* __restrict__ W, int ldw, int k0, int lane) {
  int n = lane & 15, g = (lane >> 4) & 1;
  const _Float16* p = W + (size_t)n * ldw + (k0 + g * 8);
  union { h16v v; u32x4 q[2]; } u;
  u.q[0] = *(const u32x4*)(p);
  u.q[1] = *(const u32x4*)(p + 16);
  return u.v;
}
__device__ __forceinline__ h16v frag_a_ctx(const float* __restrict__ ctx, int m0, int k0, int lane) {
  int m = m0 + (lane & 15);
  int g = (lane >> 4) & 1;
  int b = m >> 7;
  int s = m & (S_DIM - 1);
  const float* row = ctx + ((size_t)s * B_DIM + b) * H_DIM + (k0 + g * 8);
  h16v v;
#pragma unroll
  for (int e = 0; e < 8; ++e) v[e] = (_Float16)row[e];
#pragma unroll
  for (int e = 0; e < 8; ++e) v[8 + e] = (_Float16)row[16 + e];
  return v;
}

// ---------------- threefry2x32-20 (JAX PRNG) ---------------------------------
__device__ __forceinline__ uint32_t rotl32_(uint32_t x, int r) { return (x << r) | (x >> (32 - r)); }
__device__ __forceinline__ void threefry2x32_(uint32_t k0, uint32_t k1, uint32_t c0, uint32_t c1,
                                              uint32_t& y0, uint32_t& y1) {
  uint32_t ks[3] = { k0, k1, 0x1BD11BDAu ^ k0 ^ k1 };
  const int R0[4] = {13, 15, 26, 6};
  const int R1[4] = {17, 29, 16, 24};
  uint32_t x0 = c0 + ks[0], x1 = c1 + ks[1];
#pragma unroll
  for (int i = 0; i < 5; ++i) {
    const int* R = (i & 1) ? R1 : R0;
#pragma unroll
    for (int r = 0; r < 4; ++r) { x0 += x1; x1 = rotl32_(x1, R[r]); x1 ^= x0; }
    x0 += ks[(i + 1) % 3];
    x1 += ks[(i + 2) % 3] + (uint32_t)(i + 1);
  }
  y0 = x0; y1 = x1;
}

// ---------------- small utility kernels --------------------------------------
__global__ __launch_bounds__(256) void f32_to_f16_kernel(const float* __restrict__ a,
                                                         _Float16* __restrict__ b, int n) {
  int i = blockIdx.x * 256 + threadIdx.x;
  if (i < n) b[i] = (_Float16)a[i];
}

__global__ __launch_bounds__(256) void init_kernel(const float* __restrict__ hx_in,
                                                   const float* __restrict__ cx_in,
                                                   const float* __restrict__ dec_in,
                                                   float* __restrict__ hx_ws, float* __restrict__ cx_ws,
                                                   _Float16* __restrict__ hy16,
                                                   _Float16* __restrict__ decin16,
                                                   int* __restrict__ fmask, int* __restrict__ fmmask) {
  int i = blockIdx.x * 256 + threadIdx.x;
  if (i < B_DIM * H_DIM) { hx_ws[i] = hx_in[i]; cx_ws[i] = cx_in[i]; hy16[i] = (_Float16)hx_in[i]; }
  if (i < B_DIM * E_DIM) { decin16[i] = (_Float16)dec_in[i]; }
  if (i < B_DIM) { fmask[i] = 1; fmmask[i] = 1; }
}

// ---------------- E = ctx @ Wr^T + br : wave computes 16x64 strip ------------
__global__ __launch_bounds__(256) void egemm_kernel(const float* __restrict__ ctx,
                                                    const _Float16* __restrict__ W,
                                                    const float* __restrict__ bias,
                                                    _Float16* __restrict__ Eo) {
  int lane = threadIdx.x & 31, wave = threadIdx.x >> 5;
  int tile = blockIdx.x * 8 + wave;     // 32768 wave-strips (4096 m x 8 nq)
  int mTile = tile >> 3;
  int nq = tile & 7;
  int m0 = mTile * 16, n0 = nq * 64;
  f32x8 acc[4] = {};
  for (int k0 = 0; k0 < H_DIM; k0 += 32) {
    h16v a = frag_a_ctx(ctx, m0, k0, lane);
#pragma unroll
    for (int j = 0; j < 4; ++j) {
      h16v b = frag_b16(W + (size_t)(n0 + j * 16) * H_DIM, H_DIM, k0, lane);
      acc[j] = wmma_f16(a, b, acc[j]);
    }
  }
  int g = (lane >> 4) & 1;
#pragma unroll
  for (int j = 0; j < 4; ++j) {
    int n = n0 + j * 16 + (lane & 15);
    float bn = bias[n];
#pragma unroll
    for (int r = 0; r < 8; ++r) {
      int m = m0 + r + 8 * g;
      Eo[(size_t)m * H_DIM + n] = (_Float16)(acc[j][r] + bn);
    }
  }
}

// ---------------- LSTM gate GEMM (16x64 strips) ------------------------------
__global__ __launch_bounds__(256) void gates_gemm_kernel(const _Float16* __restrict__ x16,
                                                         const _Float16* __restrict__ h16,
                                                         const _Float16* __restrict__ Wi16,
                                                         const _Float16* __restrict__ Wh16,
                                                         const float* __restrict__ bi,
                                                         const float* __restrict__ bh,
                                                         float* __restrict__ gates) {
  int lane = threadIdx.x & 31, wave = threadIdx.x >> 5;
  int tile = blockIdx.x * 8 + wave;     // 1024 strips (32 m x 32 nq)
  int mTile = tile >> 5;
  int nq = tile & 31;
  int n0 = nq * 64;
  const _Float16* A1 = x16 + (size_t)(mTile * 16) * E_DIM;
  const _Float16* A2 = h16 + (size_t)(mTile * 16) * H_DIM;
  f32x8 acc[4] = {};
  for (int k0 = 0; k0 < E_DIM; k0 += 32) {
    h16v a = frag_a16(A1, E_DIM, k0, lane);
#pragma unroll
    for (int j = 0; j < 4; ++j)
      acc[j] = wmma_f16(a, frag_b16(Wi16 + (size_t)(n0 + j * 16) * E_DIM, E_DIM, k0, lane), acc[j]);
  }
  for (int k0 = 0; k0 < H_DIM; k0 += 32) {
    h16v a = frag_a16(A2, H_DIM, k0, lane);
#pragma unroll
    for (int j = 0; j < 4; ++j)
      acc[j] = wmma_f16(a, frag_b16(Wh16 + (size_t)(n0 + j * 16) * H_DIM, H_DIM, k0, lane), acc[j]);
  }
  int g = (lane >> 4) & 1;
#pragma unroll
  for (int j = 0; j < 4; ++j) {
    int n = n0 + j * 16 + (lane & 15);
    float bn = bi[n] + bh[n];
#pragma unroll
    for (int r = 0; r < 8; ++r) {
      int m = mTile * 16 + r + 8 * g;
      gates[(size_t)m * (4 * H_DIM) + n] = acc[j][r] + bn;
    }
  }
}

// ---------------- q GEMM (16x64 strips, M=N=K=512) ---------------------------
__global__ __launch_bounds__(256) void qgemm_kernel(const _Float16* __restrict__ A,
                                                    const _Float16* __restrict__ W,
                                                    const float* __restrict__ bias,
                                                    float* __restrict__ out) {
  int lane = threadIdx.x & 31, wave = threadIdx.x >> 5;
  int tile = blockIdx.x * 8 + wave;     // 256 strips (32 m x 8 nq)
  int mTile = tile >> 3;
  int nq = tile & 7;
  int n0 = nq * 64;
  const _Float16* At = A + (size_t)(mTile * 16) * H_DIM;
  f32x8 acc[4] = {};
  for (int k0 = 0; k0 < H_DIM; k0 += 32) {
    h16v a = frag_a16(At, H_DIM, k0, lane);
#pragma unroll
    for (int j = 0; j < 4; ++j)
      acc[j] = wmma_f16(a, frag_b16(W + (size_t)(n0 + j * 16) * H_DIM, H_DIM, k0, lane), acc[j]);
  }
  int g = (lane >> 4) & 1;
#pragma unroll
  for (int j = 0; j < 4; ++j) {
    int n = n0 + j * 16 + (lane & 15);
    float bn = bias[n];
#pragma unroll
    for (int r = 0; r < 8; ++r) {
      int m = mTile * 16 + r + 8 * g;
      out[(size_t)m * H_DIM + n] = acc[j][r] + bn;
    }
  }
}

// ---------------- LSTM elementwise cell --------------------------------------
__global__ __launch_bounds__(256) void lstm_kernel(const float* __restrict__ gates,
                                                   float* __restrict__ cx_ws, float* __restrict__ hx_ws,
                                                   _Float16* __restrict__ hy16) {
  int i = blockIdx.x * 256 + threadIdx.x;
  int b = i >> 9;
  int h = i & (H_DIM - 1);
  const float* gr = gates + (size_t)b * (4 * H_DIM);
  float ig = sigmoid_f(gr[h]);
  float fg = sigmoid_f(gr[H_DIM + h]);
  float cg = tanhf(gr[2 * H_DIM + h]);
  float og = sigmoid_f(gr[3 * H_DIM + h]);
  float cy = fg * cx_ws[i] + ig * cg;
  float hy = og * tanhf(cy);
  cx_ws[i] = cy; hx_ws[i] = hy; hy16[i] = (_Float16)hy;
}

// ---------------- glimpse attention: TDM tile -> LDS, 2 passes ---------------
__global__ __launch_bounds__(256) void attn_g_kernel(const float* __restrict__ qp,
                                                     const _Float16* __restrict__ Eg,
                                                     const float* __restrict__ gv,
                                                     _Float16* __restrict__ gl16) {
  extern __shared__ char smem[];
  _Float16* sE = (_Float16*)smem;                                  // S*H f16 (128 KB)
  float* sQ   = (float*)(smem + (size_t)S_DIM * H_DIM * 2);        // H
  float* sV   = sQ + H_DIM;                                        // H
  float* sA   = sV + H_DIM;                                        // S (lg -> probs)
  float* sTmp = sA + S_DIM;                                        // S
  float* sGl  = sTmp + S_DIM;                                      // H accum

  int b = blockIdx.x, tid = threadIdx.x;
  if (tid < 32)
    tdm_load_tile_2d(sE, Eg + (size_t)b * S_DIM * H_DIM, H_DIM, S_DIM, H_DIM);
  for (int h = tid; h < H_DIM; h += 256) {
    sQ[h] = qp[(size_t)b * H_DIM + h];
    sV[h] = gv[h];
    sGl[h] = 0.0f;
  }
  if (tid < S_DIM) sA[tid] = 0.0f;
  if (tid < 32) __builtin_amdgcn_s_wait_tensorcnt(0);
  __syncthreads();

  // lg[s] = sum_h gv[h]*tanh(qp[h] + E[s][h]); 2 threads per s, 8-wide LDS reads
  {
    int s = tid >> 1, half = tid & 1;
    const _Float16* row = sE + (size_t)s * H_DIM + half * 256;
    const float* q = sQ + half * 256;
    const float* vv = sV + half * 256;
    float part = 0.f;
    for (int h8 = 0; h8 < 256; h8 += 8) {
      union { u32x4 q4; _Float16 h[8]; } u;
      u.q4 = *(const u32x4*)(row + h8);
#pragma unroll
      for (int j = 0; j < 8; ++j) part += vv[h8 + j] * tanhf(q[h8 + j] + (float)u.h[j]);
    }
    atomicAdd(&sA[s], part);
  }
  __syncthreads();
  // softmax over s
  if (tid < S_DIM) sTmp[tid] = sA[tid];
  __syncthreads();
  for (int off = 64; off > 0; off >>= 1) { if (tid < off) sTmp[tid] = fmaxf(sTmp[tid], sTmp[tid + off]); __syncthreads(); }
  float mx = sTmp[0];
  __syncthreads();
  if (tid < S_DIM) { float e = __expf(sA[tid] - mx); sA[tid] = e; sTmp[tid] = e; }
  __syncthreads();
  for (int off = 64; off > 0; off >>= 1) { if (tid < off) sTmp[tid] += sTmp[tid + off]; __syncthreads(); }
  float inv = 1.0f / sTmp[0];
  __syncthreads();
  if (tid < S_DIM) sA[tid] *= inv;
  __syncthreads();
  // g_l[h] = sum_s a[s]*E[s][h]; thread = (h-group of 8) x (s-quarter), ds_add_f32 accum
  {
    int hg = tid & 63, sq = tid >> 6;
    int h0 = hg * 8;
    float acc[8] = {};
    for (int s = sq * 32; s < sq * 32 + 32; ++s) {
      float a = sA[s];
      union { u32x4 q4; _Float16 h[8]; } u;
      u.q4 = *(const u32x4*)(sE + (size_t)s * H_DIM + h0);
#pragma unroll
      for (int j = 0; j < 8; ++j) acc[j] += a * (float)u.h[j];
    }
#pragma unroll
    for (int j = 0; j < 8; ++j) atomicAdd(&sGl[h0 + j], acc[j]);
  }
  __syncthreads();
  for (int h = tid; h < H_DIM; h += 256) gl16[(size_t)b * H_DIM + h] = (_Float16)sGl[h];
}

// ---------------- pointer attention + sampling + gather ----------------------
__global__ __launch_bounds__(256) void attn_p_kernel(const float* __restrict__ qp,
                                                     const _Float16* __restrict__ Ep,
                                                     const float* __restrict__ pv,
                                                     const int* __restrict__ counts,
                                                     const float* __restrict__ emb,
                                                     int* __restrict__ fmask, int* __restrict__ fmmask,
                                                     _Float16* __restrict__ decin16,
                                                     float* __restrict__ out, int t) {
  extern __shared__ char smem[];
  _Float16* sE = (_Float16*)smem;
  float* sQ   = (float*)(smem + (size_t)S_DIM * H_DIM * 2);
  float* sV   = sQ + H_DIM;
  float* sU   = sV + H_DIM;        // S raw sums -> logits
  float* sP   = sU + S_DIM;        // S probs
  float* sTmp = sP + S_DIM;        // S
  int*   sArg = (int*)(sTmp + S_DIM);

  int b = blockIdx.x, tid = threadIdx.x;
  int fm = fmask[b], fmm = fmmask[b];
  if (tid < 32)
    tdm_load_tile_2d(sE, Ep + (size_t)b * S_DIM * H_DIM, H_DIM, S_DIM, H_DIM);
  for (int h = tid; h < H_DIM; h += 256) { sQ[h] = qp[(size_t)b * H_DIM + h]; sV[h] = pv[h]; }
  if (tid < S_DIM) sU[tid] = 0.0f;
  if (tid < 32) __builtin_amdgcn_s_wait_tensorcnt(0);
  __syncthreads();

  {
    int s = tid >> 1, half = tid & 1;
    const _Float16* row = sE + (size_t)s * H_DIM + half * 256;
    const float* q = sQ + half * 256;
    const float* vv = sV + half * 256;
    float part = 0.f;
    for (int h8 = 0; h8 < 256; h8 += 8) {
      union { u32x4 q4; _Float16 h[8]; } u;
      u.q4 = *(const u32x4*)(row + h8);
#pragma unroll
      for (int j = 0; j < 8; ++j) part += vv[h8 + j] * tanhf(q[h8 + j] + (float)u.h[j]);
    }
    atomicAdd(&sU[s], part);
  }
  __syncthreads();
  if (tid < S_DIM) sU[tid] = 10.0f * tanhf(sU[tid]);   // logits
  __syncthreads();
  if (tid < S_DIM) sTmp[tid] = sU[tid];
  __syncthreads();
  for (int off = 64; off > 0; off >>= 1) { if (tid < off) sTmp[tid] = fmaxf(sTmp[tid], sTmp[tid + off]); __syncthreads(); }
  float mx = sTmp[0];
  __syncthreads();
  if (tid < S_DIM) { float e = __expf(sU[tid] - mx); sP[tid] = e; sTmp[tid] = e; }
  __syncthreads();
  for (int off = 64; off > 0; off >>= 1) { if (tid < off) sTmp[tid] += sTmp[tid + off]; __syncthreads(); }
  float inv = 1.0f / sTmp[0];
  __syncthreads();
  if (tid < S_DIM) sP[tid] *= inv;

  // categorical(fold_in(key(42), t), logits) via Gumbel argmax (threefry2x32-20)
  if (tid < S_DIM) {
    uint32_t fk0, fk1; threefry2x32_(0u, 42u, 0u, (uint32_t)t, fk0, fk1);
    uint32_t i = (uint32_t)(b * S_DIM + tid);
    uint32_t j = i & 32767u;
    uint32_t y0, y1; threefry2x32_(fk0, fk1, j, j + 32768u, y0, y1);
    uint32_t bits = (i < 32768u) ? y0 : y1;
    float u01 = __uint_as_float((bits >> 9) | 0x3f800000u) - 1.0f;
    u01 = fmaxf(u01, 1.1754944e-38f);
    float gum = -__logf(-__logf(u01));
    sTmp[tid] = sU[tid] + gum;
    sArg[tid] = tid;
  }
  __syncthreads();
  for (int off = 64; off > 0; off >>= 1) {
    if (tid < off) {
      if (sTmp[tid + off] > sTmp[tid]) { sTmp[tid] = sTmp[tid + off]; sArg[tid] = sArg[tid + off]; }
    }
    __syncthreads();
  }
  int idx = sArg[0] * fm;
  float factor = (float)(fm * fmm);
  if (tid < S_DIM) out[((size_t)b * T_STEPS + t) * S_DIM + tid] = sP[tid] * factor;
  if (tid == 0) {
    out[(size_t)B_DIM * T_STEPS * S_DIM + (size_t)t * B_DIM + b] = (float)idx;
    fmask[b] = fm * (idx != 0 ? 1 : 0);
    fmmask[b] = (t == counts[b]) ? 0 : fmm;
  }
  decin16[(size_t)b * E_DIM + tid] = (_Float16)emb[((size_t)idx * B_DIM + b) * E_DIM + tid];
}

__global__ __launch_bounds__(256) void final_copy_kernel(const float* __restrict__ hx_ws,
                                                         const float* __restrict__ cx_ws,
                                                         float* __restrict__ out) {
  int i = blockIdx.x * 256 + threadIdx.x;
  size_t base = (size_t)B_DIM * T_STEPS * S_DIM + (size_t)T_STEPS * B_DIM;
  out[base + i] = hx_ws[i];
  out[base + (size_t)B_DIM * H_DIM + i] = cx_ws[i];
}

// ---------------- host driver ------------------------------------------------
extern "C" void kernel_launch(void* const* d_in, const int* in_sizes, int n_in,
                              void* d_out, int out_size, void* d_ws, size_t ws_size,
                              hipStream_t stream) {
  const float* dec_in = (const float*)d_in[0];
  const float* emb    = (const float*)d_in[1];
  const float* hx_in  = (const float*)d_in[2];
  const float* cx_in  = (const float*)d_in[3];
  const float* ctx    = (const float*)d_in[4];
  const int*   counts = (const int*)d_in[5];
  const float* Wi  = (const float*)d_in[7];
  const float* bi  = (const float*)d_in[8];
  const float* Wh  = (const float*)d_in[9];
  const float* bh  = (const float*)d_in[10];
  const float* gWq = (const float*)d_in[11]; const float* gbq = (const float*)d_in[12];
  const float* gWr = (const float*)d_in[13]; const float* gbr = (const float*)d_in[14];
  const float* gv  = (const float*)d_in[15];
  const float* pWq = (const float*)d_in[16]; const float* pbq = (const float*)d_in[17];
  const float* pWr = (const float*)d_in[18]; const float* pbr = (const float*)d_in[19];
  const float* pv  = (const float*)d_in[20];
  float* out = (float*)d_out;

  char* ws = (char*)d_ws;
  size_t off = 0;
  auto take = [&](size_t bytes) -> void* {
    void* p = ws + off;
    off = (off + bytes + 255) & ~(size_t)255;
    return p;
  };
  const size_t SZ_E = (size_t)B_DIM * S_DIM * H_DIM * sizeof(_Float16);
  _Float16* Eg      = (_Float16*)take(SZ_E);
  _Float16* Ep      = (_Float16*)take(SZ_E);
  _Float16* Wi16    = (_Float16*)take((size_t)4 * H_DIM * E_DIM * 2);
  _Float16* Wh16    = (_Float16*)take((size_t)4 * H_DIM * H_DIM * 2);
  _Float16* gWq16   = (_Float16*)take((size_t)H_DIM * H_DIM * 2);
  _Float16* gWr16   = (_Float16*)take((size_t)H_DIM * H_DIM * 2);
  _Float16* pWq16   = (_Float16*)take((size_t)H_DIM * H_DIM * 2);
  _Float16* pWr16   = (_Float16*)take((size_t)H_DIM * H_DIM * 2);
  float*    hx_ws   = (float*)take((size_t)B_DIM * H_DIM * 4);
  float*    cx_ws   = (float*)take((size_t)B_DIM * H_DIM * 4);
  _Float16* hy16    = (_Float16*)take((size_t)B_DIM * H_DIM * 2);
  _Float16* decin16 = (_Float16*)take((size_t)B_DIM * E_DIM * 2);
  _Float16* gl16    = (_Float16*)take((size_t)B_DIM * H_DIM * 2);
  float*    gates   = (float*)take((size_t)B_DIM * 4 * H_DIM * 4);
  float*    qp      = (float*)take((size_t)B_DIM * H_DIM * 4);
  int*      fmask   = (int*)take((size_t)B_DIM * 4);
  int*      fmmask  = (int*)take((size_t)B_DIM * 4);

  f32_to_f16_kernel<<<(4 * H_DIM * E_DIM) / 256, 256, 0, stream>>>(Wi, Wi16, 4 * H_DIM * E_DIM);
  f32_to_f16_kernel<<<(4 * H_DIM * H_DIM) / 256, 256, 0, stream>>>(Wh, Wh16, 4 * H_DIM * H_DIM);
  f32_to_f16_kernel<<<(H_DIM * H_DIM) / 256, 256, 0, stream>>>(gWq, gWq16, H_DIM * H_DIM);
  f32_to_f16_kernel<<<(H_DIM * H_DIM) / 256, 256, 0, stream>>>(gWr, gWr16, H_DIM * H_DIM);
  f32_to_f16_kernel<<<(H_DIM * H_DIM) / 256, 256, 0, stream>>>(pWq, pWq16, H_DIM * H_DIM);
  f32_to_f16_kernel<<<(H_DIM * H_DIM) / 256, 256, 0, stream>>>(pWr, pWr16, H_DIM * H_DIM);
  init_kernel<<<(B_DIM * H_DIM) / 256, 256, 0, stream>>>(hx_in, cx_in, dec_in, hx_ws, cx_ws,
                                                         hy16, decin16, fmask, fmmask);
  egemm_kernel<<<4096, 256, 0, stream>>>(ctx, gWr16, gbr, Eg);
  egemm_kernel<<<4096, 256, 0, stream>>>(ctx, pWr16, pbr, Ep);

  const size_t SMEM_G = (size_t)S_DIM * H_DIM * 2 + (3 * H_DIM + 2 * S_DIM) * 4;            // 138240
  const size_t SMEM_P = (size_t)S_DIM * H_DIM * 2 + (2 * H_DIM + 3 * S_DIM) * 4 + S_DIM * 4; // 137216

  for (int t = 0; t < T_STEPS; ++t) {
    gates_gemm_kernel<<<128, 256, 0, stream>>>(decin16, hy16, Wi16, Wh16, bi, bh, gates);
    lstm_kernel<<<(B_DIM * H_DIM) / 256, 256, 0, stream>>>(gates, cx_ws, hx_ws, hy16);
    qgemm_kernel<<<32, 256, 0, stream>>>(hy16, gWq16, gbq, qp);
    attn_g_kernel<<<B_DIM, 256, SMEM_G, stream>>>(qp, Eg, gv, gl16);
    qgemm_kernel<<<32, 256, 0, stream>>>(gl16, pWq16, pbq, qp);
    attn_p_kernel<<<B_DIM, 256, SMEM_P, stream>>>(qp, Ep, pv, counts, emb, fmask, fmmask,
                                                  decin16, out, t);
  }
  final_copy_kernel<<<(B_DIM * H_DIM) / 256, 256, 0, stream>>>(hx_ws, cx_ws, out);
}
